// HeteroGNN_64476049047925
// MI455X (gfx1250) — compile-verified, matrix operand order checked by reading, not run on hardware
//
#include <hip/hip_runtime.h>
#include <hip/hip_bf16.h>
#include <cstddef>

// ---------------- problem constants (match reference) ----------------
#define NN   10000   // nodes per type
#define EE   80000   // edges per relation
#define DIN  64      // input feature dim
#define HH   128     // hidden dim
#define DOUT 64      // output dim
#define RR   12      // relations

typedef __attribute__((ext_vector_type(16))) _Float16 v16h;
typedef __attribute__((ext_vector_type(8)))  float    v8f;

// =====================================================================
// Fragment packing. WMMA 16x16x32 f16 operand layouts (wave32):
//   A (16x32, M x K): lane<16 -> M=lane,  K in {0..7, 16..23}
//                     lane>=16 -> M=lane-16, K in {8..15, 24..31}
//   B (32x16, K x N): same K striping, N = lane & 15
// Packed form: per (tile, k-step), 32 lanes x 16 halves, contiguous per
// lane (32 B) so each wave fetches its fragment as two b128 loads.
// =====================================================================
__global__ void pack_a_f16(const float* __restrict__ A, _Float16* __restrict__ Ap,
                           int rows, int K) {
    int gid = blockIdx.x * blockDim.x + threadIdx.x;
    int KS = K >> 5;
    int total = (rows >> 4) * KS * 32;
    if (gid >= total) return;
    int lane   = gid & 31;
    int ks     = (gid >> 5) % KS;
    int m_tile = gid / (KS * 32);
    int half   = lane >> 4;
    int m      = m_tile * 16 + (lane & 15);
    v16h h;
#pragma unroll
    for (int i = 0; i < 8; ++i) {
        int k1 = ks * 32 + 8 * half + i;
        h[i]     = (_Float16)A[(size_t)m * K + k1];
        h[8 + i] = (_Float16)A[(size_t)m * K + k1 + 16];
    }
    *((v16h*)Ap + gid) = h;
}

__global__ void pack_b_f16(const float* __restrict__ W, _Float16* __restrict__ Wp,
                           int K, int Hc) {
    int gid = blockIdx.x * blockDim.x + threadIdx.x;
    int KS = K >> 5;
    int total = (Hc >> 4) * KS * 32;
    if (gid >= total) return;
    int lane   = gid & 31;
    int ks     = (gid >> 5) % KS;
    int n_tile = gid / (KS * 32);
    int half   = lane >> 4;
    int n      = n_tile * 16 + (lane & 15);
    v16h h;
#pragma unroll
    for (int i = 0; i < 8; ++i) {
        int k1 = ks * 32 + 8 * half + i;
        h[i]     = (_Float16)W[(size_t)k1 * Hc + n];
        h[8 + i] = (_Float16)W[(size_t)(k1 + 16) * Hc + n];
    }
    *((v16h*)Wp + gid) = h;
}

// =====================================================================
// WMMA GEMM on pre-packed f16 fragments: C[M,Hc] = A @ W (+ bias).
// One wave per 16x16 output tile; K/32 fully-unrolled WMMA steps, each
// fed by two b128 loads per operand. 8 waves per block.
// =====================================================================
template <int K, bool BIAS>
__global__ void gemm_wmma_packed(const _Float16* __restrict__ Ap,
                                 const _Float16* __restrict__ Wp,
                                 float* __restrict__ C,
                                 const float* __restrict__ bias,
                                 int M, int Hc) {
    constexpr int KS = K / 32;
    const int lane = threadIdx.x & 31;
    const int t    = blockIdx.x * (blockDim.x >> 5) + (threadIdx.x >> 5);
    const int nTiles = Hc >> 4;
    if (t >= (M >> 4) * nTiles) return;
    const int m_tile = t / nTiles;
    const int n_tile = t % nTiles;

    const v16h* Af = (const v16h*)Ap + (size_t)m_tile * KS * 32 + lane;
    const v16h* Bf = (const v16h*)Wp + (size_t)n_tile * KS * 32 + lane;

    v8f acc = {};
#pragma unroll
    for (int ks = 0; ks < KS; ++ks) {
        v16h av = Af[ks * 32];
        v16h bv = Bf[ks * 32];
        acc = __builtin_amdgcn_wmma_f32_16x16x32_f16(
            false, av, false, bv, (short)0, acc, false, false);
    }

    // C layout: VGPR i -> M = i + 8*(lane>=16), N = lane & 15.
    const int half = lane >> 4;
    const int n    = n_tile * 16 + (lane & 15);
    float badd = 0.f;
    if (BIAS) badd = bias[n];
#pragma unroll
    for (int i = 0; i < 8; ++i) {
        int m = m_tile * 16 + i + 8 * half;
        C[(size_t)m * Hc + n] = acc[i] + badd;
    }
}

// =====================================================================
// out[n] = sum_h X[n,h] * att[h]   (one wave per row)
// =====================================================================
__global__ void rowdot(const float* __restrict__ X, const float* __restrict__ att,
                       float* __restrict__ out, int nrows, int Hc) {
    int row  = blockIdx.x * (blockDim.x >> 5) + (threadIdx.x >> 5);
    int lane = threadIdx.x & 31;
    if (row >= nrows) return;
    float s = 0.f;
    for (int h = lane; h < Hc; h += 32) s += X[(size_t)row * Hc + h] * att[h];
#pragma unroll
    for (int off = 16; off > 0; off >>= 1) s += __shfl_down(s, off, 32);
    if (lane == 0) out[row] = s;
}

// HeteroConv 'sum' bias: Xnext[t][n][h] = sum of b[r][h] over the 3
// relations r whose destination type is t.
__global__ void bias_init(float* __restrict__ X, const float* __restrict__ b) {
    int idx = blockIdx.x * blockDim.x + threadIdx.x;
    if (idx >= 4 * NN * HH) return;
    int t = idx / (NN * HH);
    int h = idx % HH;
    const int rel[4][3] = {{3, 6, 9}, {0, 7, 10}, {1, 4, 11}, {2, 5, 8}};
    X[idx] = b[rel[t][0] * HH + h] + b[rel[t][1] * HH + h] + b[rel[t][2] * HH + h];
}

__global__ void init_maxden(unsigned* __restrict__ umax, float* __restrict__ den, int n) {
    int i = blockIdx.x * blockDim.x + threadIdx.x;
    if (i < n) { umax[i] = 0u; den[i] = 0.f; }
}

// Pass 1: leaky-relu edge score + segment-max via order-preserving uint key.
__global__ void edge_scores(const int* __restrict__ src, const int* __restrict__ dst,
                            const float* __restrict__ a_s, const float* __restrict__ a_d,
                            float* __restrict__ ebuf, unsigned* __restrict__ umax, int nE) {
    int e = blockIdx.x * blockDim.x + threadIdx.x;
    if (e >= nE) return;
    float v = a_s[src[e]] + a_d[dst[e]];
    v = (v > 0.f) ? v : 0.2f * v;
    ebuf[e] = v;
    unsigned b = __float_as_uint(v);
    unsigned key = (b & 0x80000000u) ? ~b : (b | 0x80000000u);
    atomicMax(&umax[dst[e]], key);
}

// Pass 2: w = exp(e - max[dst]); segment-sum of w.
__global__ void edge_expsum(const int* __restrict__ dst, float* __restrict__ ebuf,
                            const unsigned* __restrict__ umax, float* __restrict__ den, int nE) {
    int e = blockIdx.x * blockDim.x + threadIdx.x;
    if (e >= nE) return;
    unsigned u = umax[dst[e]];
    float m = (u & 0x80000000u) ? __uint_as_float(u & 0x7FFFFFFFu) : __uint_as_float(~u);
    float w = __expf(ebuf[e] - m);
    ebuf[e] = w;
    atomicAdd(&den[dst[e]], w);
}

// Pass 3: scatter alpha * xs[src] into the destination-type accumulator.
__global__ void edge_scatter(const int* __restrict__ src, const int* __restrict__ dst,
                             const float* __restrict__ ebuf, const float* __restrict__ den,
                             const float* __restrict__ xs, float* __restrict__ outT, int nE) {
    int e = blockIdx.x;
    if (e >= nE) return;
    int s = src[e], d = dst[e];
    float alpha = ebuf[e] / den[d];
    int h = threadIdx.x;
    atomicAdd(&outT[(size_t)d * HH + h], alpha * xs[(size_t)s * HH + h]);
}

__global__ void relu_inplace(float* __restrict__ X, int n) {
    int i = blockIdx.x * blockDim.x + threadIdx.x;
    if (i < n) { float v = X[i]; X[i] = v > 0.f ? v : 0.f; }
}

// =====================================================================
// Host orchestration
// =====================================================================
extern "C" void kernel_launch(void* const* d_in, const int* in_sizes, int n_in,
                              void* d_out, int out_size, void* d_ws, size_t ws_size,
                              hipStream_t stream) {
    (void)in_sizes; (void)n_in; (void)out_size; (void)ws_size;

    const float* x_state      = (const float*)d_in[0];
    const float* x_action     = (const float*)d_in[1];
    const float* x_background = (const float*)d_in[2];
    const float* x_t          = (const float*)d_in[3];
    const int*   edge_index   = (const int*)d_in[4];
    const float* Wsrc0 = (const float*)d_in[5];
    const float* Wdst0 = (const float*)d_in[6];
    const float* asrc0 = (const float*)d_in[7];
    const float* adst0 = (const float*)d_in[8];
    const float* b0    = (const float*)d_in[9];
    const float* Wsrc1 = (const float*)d_in[10];
    const float* Wdst1 = (const float*)d_in[11];
    const float* asrc1 = (const float*)d_in[12];
    const float* adst1 = (const float*)d_in[13];
    const float* b1    = (const float*)d_in[14];
    const float* lin_w = (const float*)d_in[15];
    const float* lin_b = (const float*)d_in[16];

    static const int SRC_T[RR] = {0, 0, 0, 1, 1, 1, 2, 2, 2, 3, 3, 3};
    static const int DST_T[RR] = {1, 2, 3, 0, 2, 3, 0, 1, 3, 0, 1, 2};

    // ---- workspace carve-up (~62 MB) ----
    float*    X1   = (float*)d_ws;                 // [4][N][H] layer-1 output
    float*    X2   = X1 + (size_t)4 * NN * HH;     // [4][N][H] layer-2 output
    float*    xs   = X2 + (size_t)4 * NN * HH;     // [N][H] per-relation src proj
    float*    xd   = xs + (size_t)NN * HH;         // [N][H] per-relation dst proj
    float*    a_s  = xd + (size_t)NN * HH;         // [N]
    float*    a_d  = a_s + NN;                     // [N]
    float*    den  = a_d + NN;                     // [N]
    unsigned* umax = (unsigned*)(den + NN);        // [N]
    float*    ebuf = (float*)(umax + NN);          // [E]
    // packed f16 regions (each type gets an N x 128 slot regardless of K)
    _Float16* ApBase = (_Float16*)(ebuf + EE);                     // 4 * N * 128 halves
    _Float16* WpS    = ApBase + (size_t)4 * NN * HH;               // K*H <= 128*128 halves
    _Float16* WpD    = WpS + (size_t)HH * HH;
    _Float16* WpLin  = WpD + (size_t)HH * HH;                      // 128*64 halves

    const int bStd = 256;
    const int gDot  = (NN + 3) / 4;
    const int gN    = (NN + 255) / 256;
    const int gE    = (EE + 255) / 256;
    const int gBias = (4 * NN * HH + 255) / 256;

    const int tilesH   = (NN / 16) * (HH / 16);    // 5000 tiles
    const int tilesO   = (NN / 16) * (DOUT / 16);  // 2500 tiles
    const int gGemmH   = (tilesH + 7) / 8;         // 8 waves / block
    const int gGemmO   = (tilesO + 7) / 8;
    const int packA0_n = (NN / 16) * (DIN / 32) * 32;   // threads per type, K=64
    const int packA1_n = (NN / 16) * (HH / 32) * 32;    // threads per type, K=128
    const int packB0_n = (HH / 16) * (DIN / 32) * 32;
    const int packB1_n = (HH / 16) * (HH / 32) * 32;
    const int packBL_n = (DOUT / 16) * (HH / 32) * 32;

    const float* Xin0[4] = {x_state, x_action, x_background, x_t};

    // ---------------- layer 0 (K = DIN = 64) ----------------
    for (int t = 0; t < 4; ++t)
        pack_a_f16<<<(packA0_n + bStd - 1) / bStd, bStd, 0, stream>>>(
            Xin0[t], ApBase + (size_t)t * NN * HH, NN, DIN);
    bias_init<<<gBias, bStd, 0, stream>>>(X1, b0);

    for (int r = 0; r < RR; ++r) {
        const int st = SRC_T[r], dt = DST_T[r];
        const int* ei  = edge_index + (size_t)r * 2 * EE;
        const int* esr = ei;
        const int* eds = ei + EE;

        pack_b_f16<<<(packB0_n + bStd - 1) / bStd, bStd, 0, stream>>>(
            Wsrc0 + (size_t)r * DIN * HH, WpS, DIN, HH);
        pack_b_f16<<<(packB0_n + bStd - 1) / bStd, bStd, 0, stream>>>(
            Wdst0 + (size_t)r * DIN * HH, WpD, DIN, HH);
        gemm_wmma_packed<DIN, false><<<gGemmH, bStd, 0, stream>>>(
            ApBase + (size_t)st * NN * HH, WpS, xs, nullptr, NN, HH);
        gemm_wmma_packed<DIN, false><<<gGemmH, bStd, 0, stream>>>(
            ApBase + (size_t)dt * NN * HH, WpD, xd, nullptr, NN, HH);
        rowdot<<<gDot, 128, 0, stream>>>(xs, asrc0 + (size_t)r * HH, a_s, NN, HH);
        rowdot<<<gDot, 128, 0, stream>>>(xd, adst0 + (size_t)r * HH, a_d, NN, HH);
        init_maxden<<<gN, bStd, 0, stream>>>(umax, den, NN);
        edge_scores<<<gE, bStd, 0, stream>>>(esr, eds, a_s, a_d, ebuf, umax, EE);
        edge_expsum<<<gE, bStd, 0, stream>>>(eds, ebuf, umax, den, EE);
        edge_scatter<<<EE, HH, 0, stream>>>(esr, eds, ebuf, den, xs,
                                            X1 + (size_t)dt * NN * HH, EE);
    }
    relu_inplace<<<gBias, bStd, 0, stream>>>(X1, 4 * NN * HH);

    // ---------------- layer 1 (K = H = 128) ----------------
    for (int t = 0; t < 4; ++t)
        pack_a_f16<<<(packA1_n + bStd - 1) / bStd, bStd, 0, stream>>>(
            X1 + (size_t)t * NN * HH, ApBase + (size_t)t * NN * HH, NN, HH);
    bias_init<<<gBias, bStd, 0, stream>>>(X2, b1);

    for (int r = 0; r < RR; ++r) {
        const int st = SRC_T[r], dt = DST_T[r];
        const int* ei  = edge_index + (size_t)r * 2 * EE;
        const int* esr = ei;
        const int* eds = ei + EE;

        pack_b_f16<<<(packB1_n + bStd - 1) / bStd, bStd, 0, stream>>>(
            Wsrc1 + (size_t)r * HH * HH, WpS, HH, HH);
        pack_b_f16<<<(packB1_n + bStd - 1) / bStd, bStd, 0, stream>>>(
            Wdst1 + (size_t)r * HH * HH, WpD, HH, HH);
        gemm_wmma_packed<HH, false><<<gGemmH, bStd, 0, stream>>>(
            ApBase + (size_t)st * NN * HH, WpS, xs, nullptr, NN, HH);
        gemm_wmma_packed<HH, false><<<gGemmH, bStd, 0, stream>>>(
            ApBase + (size_t)dt * NN * HH, WpD, xd, nullptr, NN, HH);
        rowdot<<<gDot, 128, 0, stream>>>(xs, asrc1 + (size_t)r * HH, a_s, NN, HH);
        rowdot<<<gDot, 128, 0, stream>>>(xd, adst1 + (size_t)r * HH, a_d, NN, HH);
        init_maxden<<<gN, bStd, 0, stream>>>(umax, den, NN);
        edge_scores<<<gE, bStd, 0, stream>>>(esr, eds, a_s, a_d, ebuf, umax, EE);
        edge_expsum<<<gE, bStd, 0, stream>>>(eds, ebuf, umax, den, EE);
        edge_scatter<<<EE, HH, 0, stream>>>(esr, eds, ebuf, den, xs,
                                            X2 + (size_t)dt * NN * HH, EE);
    }
    relu_inplace<<<gBias, bStd, 0, stream>>>(X2, 4 * NN * HH);

    // ---------------- final linear on 'state' ----------------
    pack_a_f16<<<(packA1_n + bStd - 1) / bStd, bStd, 0, stream>>>(
        X2 /* state */, ApBase, NN, HH);
    pack_b_f16<<<(packBL_n + bStd - 1) / bStd, bStd, 0, stream>>>(
        lin_w, WpLin, HH, DOUT);
    gemm_wmma_packed<HH, true><<<gGemmO, bStd, 0, stream>>>(
        ApBase, WpLin, (float*)d_out, lin_b, NN, DOUT);
}
